// TransformerBlock_34419867910997
// MI455X (gfx1250) — compile-verified, matrix operand order checked by reading, not run on hardware
//
#include <hip/hip_runtime.h>
#include <math.h>

typedef unsigned int  u32;
typedef unsigned short u16;

typedef __bf16 v16bf __attribute__((ext_vector_type(16)));
typedef float  v8f   __attribute__((ext_vector_type(8)));

union BF16Frag { v16bf v; u32 u[8]; };

#define C_DIM 192
#define T_DIM 10
#define H_DIM 64
#define W_DIM 64
#define NSP   (T_DIM * H_DIM * W_DIM)   // 40960
#define O_QKV 576
#define HF    510
#define HF2P  1024
#define HFP   512
#define HEADS 4
#define CH    48

__device__ __forceinline__ u16 f2bf(float f) {
    u32 u = __float_as_uint(f);
    u32 r = (u + 0x7FFFu + ((u >> 16) & 1u)) >> 16;   // RNE
    return (u16)r;
}
__device__ __forceinline__ float bf2f(u16 h) {
    return __uint_as_float(((u32)h) << 16);
}
// K offset held by VGPR v of an A/B bf16 fragment for this lane half (kb = 0 or 8)
__device__ __forceinline__ int k0_of(int v, int kb) {
    return (v < 4 ? 2 * v : 16 + 2 * (v - 4)) + kb;
}

// ---------------------------------------------------------------------------
// prep: weights -> bf16 (with padding), zero Gram accumulator + g pad rows
// ---------------------------------------------------------------------------
__global__ void prep_kernel(const float* __restrict__ qkv_w, const float* __restrict__ proj_w,
                            const float* __restrict__ pin_w, const float* __restrict__ pout_w,
                            u16* __restrict__ wq, u16* __restrict__ wp, u16* __restrict__ wi,
                            u16* __restrict__ wo, float* __restrict__ G, u16* __restrict__ gpad) {
    long i = (long)blockIdx.x * blockDim.x + threadIdx.x;
    const long N_wq = 576L * 192, N_wp = 192L * 192, N_wi = 1024L * 192;
    const long N_wo = 192L * 512, N_G = 4L * 48 * 48, N_gp = 2L * NSP;
    if (i < N_wq) { wq[i] = f2bf(qkv_w[i]); return; } i -= N_wq;
    if (i < N_wp) { wp[i] = f2bf(proj_w[i]); return; } i -= N_wp;
    if (i < N_wi) { long r = i / 192; wi[i] = (r < 1020) ? f2bf(pin_w[i]) : (u16)0; return; } i -= N_wi;
    if (i < N_wo) { long r = i / 512, c = i % 512;
                    wo[i] = (c < HF) ? f2bf(pout_w[r * HF + c]) : (u16)0; return; } i -= N_wo;
    if (i < N_G)  { G[i] = 0.f; return; } i -= N_G;
    if (i < N_gp) { gpad[i] = 0; }
}

// ---------------------------------------------------------------------------
// LayerNorm over channels, fp32 in -> bf16 out, [C][N] layout
// ---------------------------------------------------------------------------
__global__ __launch_bounds__(256) void ln_kernel(const float* __restrict__ x,
                                                 const float* __restrict__ w,
                                                 const float* __restrict__ b,
                                                 u16* __restrict__ out) {
    int n = blockIdx.x * 256 + threadIdx.x;
    if (n >= NSP) return;
    float s = 0.f, ss = 0.f;
    for (int c = 0; c < C_DIM; ++c) { float v = x[c * NSP + n]; s += v; ss += v * v; }
    float mean = s * (1.0f / C_DIM);
    float var  = ss * (1.0f / C_DIM) - mean * mean;
    float rstd = rsqrtf(var + 1e-5f);
    for (int c = 0; c < C_DIM; ++c) {
        float v = (x[c * NSP + n] - mean) * rstd * w[c] + b[c];
        out[c * NSP + n] = f2bf(v);
    }
}

// ---------------------------------------------------------------------------
// Generic WMMA bf16 GEMM: Y[M][NSP] = W[M][K] * X[K][NSP] (+ residual)
// block = 256 threads (8 waves), tile 64M x 128N, K step 32
// grid: (NSP/128, M/64)
// ---------------------------------------------------------------------------
template <bool OUT_BF16, bool RESID>
__global__ __launch_bounds__(256) void gemm_bf16(const u16* __restrict__ W,
                                                 const u16* __restrict__ X,
                                                 void* __restrict__ Y,
                                                 const float* __restrict__ resid,
                                                 int K) {
    __shared__ u16 As[64 * 36];     // [m][k], row stride 36 (pairs contiguous)
    __shared__ u32 Bs[16 * 128];    // [kpair][n] packed (k, k+1) bf16 pairs

    const int tid  = threadIdx.x;
    const int wave = tid >> 5, lane = tid & 31;
    const int wm = wave >> 1, wn = wave & 1;
    const int lm = lane & 15, kb = (lane >> 4) * 8;
    const int mBase = blockIdx.y * 64;
    const int nBase = blockIdx.x * 128;

    v8f acc[4] = {v8f{}, v8f{}, v8f{}, v8f{}};

    for (int k0 = 0; k0 < K; k0 += 32) {
        // ---- stage A: 64x32 bf16 weights
        {
            int r = tid >> 2;
            int c = (tid & 3) * 8;
            uint4 d = *(const uint4*)(W + (long)(mBase + r) * K + k0 + c);
            u32* dst = (u32*)&As[r * 36 + c];
            ((uint2*)dst)[0] = make_uint2(d.x, d.y);
            ((uint2*)dst)[1] = make_uint2(d.z, d.w);
        }
        // ---- stage B: 32x128 bf16 activations, packed as k-pairs
        {
            int kk = tid >> 4;
            int n0 = (tid & 15) * 8;
            const u16* s0 = X + (long)(k0 + 2 * kk) * NSP + nBase + n0;
            uint4 a = *(const uint4*)s0;
            uint4 b = *(const uint4*)(s0 + NSP);
            u32* dst = &Bs[kk * 128 + n0];
            uint4 o0, o1;
            o0.x = (a.x & 0xFFFFu) | (b.x << 16); o0.y = (a.x >> 16) | (b.x & 0xFFFF0000u);
            o0.z = (a.y & 0xFFFFu) | (b.y << 16); o0.w = (a.y >> 16) | (b.y & 0xFFFF0000u);
            o1.x = (a.z & 0xFFFFu) | (b.z << 16); o1.y = (a.z >> 16) | (b.z & 0xFFFF0000u);
            o1.z = (a.w & 0xFFFFu) | (b.w << 16); o1.w = (a.w >> 16) | (b.w & 0xFFFF0000u);
            *(uint4*)dst = o0; *(uint4*)(dst + 4) = o1;
        }
        __syncthreads();

        // ---- fragments + WMMA
        BF16Frag af;
        const int am = wm * 16 + lm;
        #pragma unroll
        for (int v = 0; v < 8; ++v)
            af.u[v] = *(const u32*)&As[am * 36 + k0_of(v, kb)];

        #pragma unroll
        for (int t = 0; t < 4; ++t) {
            BF16Frag bfv;
            const int bn = wn * 64 + t * 16 + lm;
            const int kph = kb >> 1;    // 0 or 4
            #pragma unroll
            for (int v = 0; v < 8; ++v) {
                int kp = (v < 4 ? v : 8 + (v - 4)) + kph;   // k0_of(v,kb)/2
                bfv.u[v] = Bs[kp * 128 + bn];
            }
            acc[t] = __builtin_amdgcn_wmma_f32_16x16x32_bf16(
                false, af.v, false, bfv.v, (short)0, acc[t], false, false);
        }
        __syncthreads();
    }

    // ---- store (vgpr v: m_local = v + 8*(lane>=16), n_local = lane&15)
    const int mloc = (lane >> 4) << 3;
    #pragma unroll
    for (int t = 0; t < 4; ++t) {
        #pragma unroll
        for (int v = 0; v < 8; ++v) {
            long m = mBase + wm * 16 + mloc + v;
            long n = nBase + wn * 64 + t * 16 + lm;
            float val = acc[t][v];
            if constexpr (RESID) val += resid[m * NSP + n];
            if constexpr (OUT_BF16) ((u16*)Y)[m * NSP + n] = f2bf(val);
            else                    ((float*)Y)[m * NSP + n] = val;
        }
    }
}

// ---------------------------------------------------------------------------
// Depthwise 3x3x3 conv, SAME padding. bf16 in [C][T][H][W] -> fp32 out.
// grid: (16 spatial tiles, T, channels); block 256 = 16x16 tile
// ---------------------------------------------------------------------------
__global__ __launch_bounds__(256) void dwconv_k(const u16* __restrict__ in,
                                                const float* __restrict__ wdw,
                                                float* __restrict__ outp) {
    const int ch = blockIdx.z, t = blockIdx.y, tile = blockIdx.x;
    const int ty0 = (tile >> 2) * 16, tx0 = (tile & 3) * 16;
    __shared__ u16 sm[3][18][18];
    const int tid = threadIdx.x;
    for (int i = tid; i < 3 * 18 * 18; i += 256) {
        int p = i / 324, rem = i % 324, yy = rem / 18, xx = rem % 18;
        int gt = t + p - 1, gy = ty0 + yy - 1, gx = tx0 + xx - 1;
        u16 v = 0;
        if (gt >= 0 && gt < T_DIM && (unsigned)gy < H_DIM && (unsigned)gx < W_DIM)
            v = in[(((long)ch * T_DIM + gt) * H_DIM + gy) * W_DIM + gx];
        sm[p][yy][xx] = v;
    }
    __syncthreads();
    const int ly = tid >> 4, lx = tid & 15;
    float acc = 0.f;
    #pragma unroll
    for (int p = 0; p < 3; ++p)
        #pragma unroll
        for (int dy = 0; dy < 3; ++dy)
            #pragma unroll
            for (int dx = 0; dx < 3; ++dx)
                acc += bf2f(sm[p][ly + dy][lx + dx]) * wdw[ch * 27 + (p * 3 + dy) * 3 + dx];
    outp[(((long)ch * T_DIM + t) * H_DIM + ty0 + ly) * W_DIM + tx0 + lx] = acc;
}

// ---------------------------------------------------------------------------
// Per-channel reciprocal L2 norms of q (rows 0..191) and k (rows 192..383)
// ---------------------------------------------------------------------------
__global__ __launch_bounds__(256) void qk_norms(const float* __restrict__ qkvd,
                                                float* __restrict__ rq, float* __restrict__ rk) {
    const int row = blockIdx.x;              // 0..383
    const float* p = qkvd + (long)row * NSP;
    float ss = 0.f;
    for (int i = threadIdx.x; i < NSP; i += 256) { float v = p[i]; ss += v * v; }
    __shared__ float red[256];
    red[threadIdx.x] = ss; __syncthreads();
    for (int s = 128; s > 0; s >>= 1) {
        if (threadIdx.x < s) red[threadIdx.x] += red[threadIdx.x + s];
        __syncthreads();
    }
    if (threadIdx.x == 0) {
        float r = 1.0f / fmaxf(sqrtf(red[0]), 1e-12f);
        if (row < 192) rq[row] = r; else rk[row - 192] = r;
    }
}

// ---------------------------------------------------------------------------
// Gram matrix per head: G[h][c][d] += sum_n q[c,n]*k[d,n]  (WMMA over K=n)
// grid: (16 n-chunks, 4 heads), block 288 = 9 waves, one 16x16 tile per wave
// (no lane-divergent WMMA: every wave's tile assignment is wave-uniform)
// ---------------------------------------------------------------------------
__global__ __launch_bounds__(288) void gram_wmma(const float* __restrict__ qkvd,
                                                 float* __restrict__ G) {
    const int h = blockIdx.y, chunk = blockIdx.x;
    const int nPer = NSP / 16;               // 2560
    const int nStart = chunk * nPer;
    __shared__ u16 Qs[48 * 132];
    __shared__ u16 Ks[48 * 132];
    const int tid = threadIdx.x, wave = tid >> 5, lane = tid & 31;
    const int lm = lane & 15, kb = (lane >> 4) * 8;
    const int mt0 = wave / 3, nt0 = wave % 3;   // 9 waves -> 9 tiles of 48x48

    v8f acc = v8f{};

    for (int nb = 0; nb < nPer; nb += 128) {
        const int nBase = nStart + nb;
        for (int i = tid; i < 48 * 32; i += 288) {
            int r = i >> 5, c4 = (i & 31) * 4;
            float4 q4 = *(const float4*)(qkvd + ((long)(h * CH + r)) * NSP + nBase + c4);
            float4 k4 = *(const float4*)(qkvd + ((long)(192 + h * CH + r)) * NSP + nBase + c4);
            u16* qd = &Qs[r * 132 + c4];
            u16* kd = &Ks[r * 132 + c4];
            qd[0] = f2bf(q4.x); qd[1] = f2bf(q4.y); qd[2] = f2bf(q4.z); qd[3] = f2bf(q4.w);
            kd[0] = f2bf(k4.x); kd[1] = f2bf(k4.y); kd[2] = f2bf(k4.z); kd[3] = f2bf(k4.w);
        }
        __syncthreads();
        #pragma unroll
        for (int ks = 0; ks < 128; ks += 32) {
            BF16Frag af, bfk;
            #pragma unroll
            for (int v = 0; v < 8; ++v) {
                int kk = ks + k0_of(v, kb);
                af.u[v]  = *(const u32*)&Qs[(mt0 * 16 + lm) * 132 + kk];
                bfk.u[v] = *(const u32*)&Ks[(nt0 * 16 + lm) * 132 + kk];
            }
            acc = __builtin_amdgcn_wmma_f32_16x16x32_bf16(
                false, af.v, false, bfk.v, (short)0, acc, false, false);
        }
        __syncthreads();
    }

    const int mloc = (lane >> 4) << 3;
    #pragma unroll
    for (int v = 0; v < 8; ++v)
        atomicAdd(&G[((long)h * CH + mt0 * 16 + mloc + v) * CH + nt0 * 16 + lm], acc[v]);
}

// ---------------------------------------------------------------------------
// Scale (rq*rk*temperature) + softmax over d; attn bf16 [4][48][64], cols 48..63 = 0
// ---------------------------------------------------------------------------
__global__ void softmax_attn(const float* __restrict__ G, const float* __restrict__ rq,
                             const float* __restrict__ rk, const float* __restrict__ temp,
                             u16* __restrict__ attn) {
    int r = threadIdx.x;
    if (r >= 192) return;
    int h = r / CH, c = r % CH;
    float tq = rq[h * CH + c] * temp[h];
    float vals[CH];
    float mx = -1e30f;
    for (int d = 0; d < CH; ++d) {
        float v = G[((long)h * CH + c) * CH + d] * tq * rk[h * CH + d];
        vals[d] = v; mx = fmaxf(mx, v);
    }
    float s = 0.f;
    for (int d = 0; d < CH; ++d) { vals[d] = expf(vals[d] - mx); s += vals[d]; }
    float inv = 1.0f / s;
    u16* dst = attn + ((long)h * CH + c) * 64;
    for (int d = 0; d < CH; ++d) dst[d] = f2bf(vals[d] * inv);
    for (int d = CH; d < 64; ++d) dst[d] = 0;
}

// ---------------------------------------------------------------------------
// out[h*48+c, n] = sum_d attn[h][c][d] * v[d][n]   (WMMA, K padded to 64)
// grid: (NSP/64, 4 heads), block 96 (3 waves, one 16-row slab each)
// ---------------------------------------------------------------------------
__global__ __launch_bounds__(96) void attn_v(const u16* __restrict__ attn,
                                             const float* __restrict__ qkvd,
                                             u16* __restrict__ outp) {
    const int h = blockIdx.y;
    const int n0 = blockIdx.x * 64;
    const int wave = threadIdx.x >> 5, lane = threadIdx.x & 31;
    const int lm = lane & 15, kb = (lane >> 4) * 8;
    const int mt = wave;     // 0..2
    v8f acc[4] = {v8f{}, v8f{}, v8f{}, v8f{}};
    const u16* arow = attn + ((long)h * CH + mt * 16 + lm) * 64;

    #pragma unroll
    for (int ks = 0; ks < 64; ks += 32) {
        BF16Frag af;
        #pragma unroll
        for (int v = 0; v < 8; ++v)
            af.u[v] = *(const u32*)&arow[ks + k0_of(v, kb)];
        #pragma unroll
        for (int t = 0; t < 4; ++t) {
            BF16Frag bfv;
            const int n = n0 + t * 16 + lm;
            #pragma unroll
            for (int v = 0; v < 8; ++v) {
                int k = ks + k0_of(v, kb);
                int r0 = 384 + h * CH + k;     if (r0 > 575) r0 = 575;   // attn pad is 0
                int r1 = 384 + h * CH + k + 1; if (r1 > 575) r1 = 575;
                float lo = qkvd[(long)r0 * NSP + n];
                float hi = qkvd[(long)r1 * NSP + n];
                bfv.u[v] = (u32)f2bf(lo) | ((u32)f2bf(hi) << 16);
            }
            acc[t] = __builtin_amdgcn_wmma_f32_16x16x32_bf16(
                false, af.v, false, bfv.v, (short)0, acc[t], false, false);
        }
    }
    const int mloc = (lane >> 4) << 3;
    #pragma unroll
    for (int t = 0; t < 4; ++t)
        #pragma unroll
        for (int v = 0; v < 8; ++v)
            outp[((long)h * CH + mt * 16 + mloc + v) * NSP + n0 + t * 16 + lm] = f2bf(acc[t][v]);
}

// ---------------------------------------------------------------------------
// FFN gate: g[c] = gelu(dw(h,c)) * dw(h,c+510), bf16 out [512][N]
// grid: (16, T, 510), block 256
// ---------------------------------------------------------------------------
__global__ __launch_bounds__(256) void ffn_gate(const u16* __restrict__ hbuf,
                                                const float* __restrict__ wdw,
                                                u16* __restrict__ g) {
    const int c = blockIdx.z, t = blockIdx.y, tile = blockIdx.x;
    const int ty0 = (tile >> 2) * 16, tx0 = (tile & 3) * 16;
    __shared__ u16 sm1[3][18][18], sm2[3][18][18];
    const int tid = threadIdx.x;
    for (int i = tid; i < 3 * 18 * 18; i += 256) {
        int p = i / 324, rem = i % 324, yy = rem / 18, xx = rem % 18;
        int gt = t + p - 1, gy = ty0 + yy - 1, gx = tx0 + xx - 1;
        u16 v1 = 0, v2 = 0;
        if (gt >= 0 && gt < T_DIM && (unsigned)gy < H_DIM && (unsigned)gx < W_DIM) {
            long base = ((long)gt * H_DIM + gy) * W_DIM + gx;
            v1 = hbuf[(long)c * NSP + base];
            v2 = hbuf[(long)(c + HF) * NSP + base];
        }
        sm1[p][yy][xx] = v1;
        sm2[p][yy][xx] = v2;
    }
    __syncthreads();
    const int ly = tid >> 4, lx = tid & 15;
    float a1 = 0.f, a2 = 0.f;
    #pragma unroll
    for (int p = 0; p < 3; ++p)
        #pragma unroll
        for (int dy = 0; dy < 3; ++dy)
            #pragma unroll
            for (int dx = 0; dx < 3; ++dx) {
                float w1 = wdw[c * 27 + (p * 3 + dy) * 3 + dx];
                float w2 = wdw[(c + HF) * 27 + (p * 3 + dy) * 3 + dx];
                a1 += bf2f(sm1[p][ly + dy][lx + dx]) * w1;
                a2 += bf2f(sm2[p][ly + dy][lx + dx]) * w2;
            }
    float ge = 0.5f * a1 * (1.0f + erff(a1 * 0.70710678118654752f));
    g[(long)c * NSP + ((long)t * H_DIM + ty0 + ly) * W_DIM + tx0 + lx] = f2bf(ge * a2);
}

// ---------------------------------------------------------------------------
extern "C" void kernel_launch(void* const* d_in, const int* in_sizes, int n_in,
                              void* d_out, int out_size, void* d_ws, size_t ws_size,
                              hipStream_t stream) {
    (void)in_sizes; (void)n_in; (void)out_size; (void)ws_size;
    const float* x      = (const float*)d_in[0];
    const float* ln1_w  = (const float*)d_in[1];
    const float* ln1_b  = (const float*)d_in[2];
    const float* qkv_w  = (const float*)d_in[3];
    const float* qkv_dw = (const float*)d_in[4];
    const float* proj_w = (const float*)d_in[5];
    const float* temp   = (const float*)d_in[6];
    const float* ln2_w  = (const float*)d_in[7];
    const float* ln2_b  = (const float*)d_in[8];
    const float* pin_w  = (const float*)d_in[9];
    const float* ffn_dw = (const float*)d_in[10];
    const float* pout_w = (const float*)d_in[11];
    float* outp = (float*)d_out;

    char* w = (char*)d_ws;
    size_t off = 0;
    auto take = [&](size_t bytes) -> char* {
        char* p = w + off;
        off += (bytes + 255) & ~(size_t)255;
        return p;
    };
    u16*   wq    = (u16*)  take(576L * 192 * 2);
    u16*   wp    = (u16*)  take(192L * 192 * 2);
    u16*   wi    = (u16*)  take(1024L * 192 * 2);
    u16*   wo    = (u16*)  take(192L * 512 * 2);
    float* G     = (float*)take(4L * 48 * 48 * 4);
    float* rq    = (float*)take(192 * 4);
    float* rk    = (float*)take(192 * 4);
    u16*   attn  = (u16*)  take(4L * 48 * 64 * 2);
    u16*   lnb   = (u16*)  take((size_t)192 * NSP * 2);
    u16*   qkvl  = (u16*)  take((size_t)576 * NSP * 2);
    float* qkvd  = (float*)take((size_t)576 * NSP * 4);
    u16*   gbuf  = (u16*)  take((size_t)512 * NSP * 2);
    // aliases (lifetimes are disjoint along the pipeline):
    u16*   aout  = qkvl;                                          // bf16 [192][N]
    float* ybuf  = (float*)((char*)qkvl + (size_t)192 * NSP * 2); // fp32 [192][N]
    u16*   hbuf  = (u16*)qkvd;                                    // bf16 [1024][N]

    const long prepN = 576L*192 + 192L*192 + 1024L*192 + 192L*512 + 4L*48*48 + 2L*NSP;
    prep_kernel<<<(unsigned)((prepN + 255) / 256), 256, 0, stream>>>(
        qkv_w, proj_w, pin_w, pout_w, wq, wp, wi, wo, G, gbuf + (size_t)HF * NSP);

    // ---- attention branch
    ln_kernel<<<NSP / 256, 256, 0, stream>>>(x, ln1_w, ln1_b, lnb);
    gemm_bf16<true, false><<<dim3(NSP / 128, O_QKV / 64), 256, 0, stream>>>(
        wq, lnb, (void*)qkvl, nullptr, 192);
    dwconv_k<<<dim3(16, T_DIM, O_QKV), 256, 0, stream>>>(qkvl, qkv_dw, qkvd);
    qk_norms<<<384, 256, 0, stream>>>(qkvd, rq, rk);
    gram_wmma<<<dim3(16, HEADS), 288, 0, stream>>>(qkvd, G);
    softmax_attn<<<1, 192, 0, stream>>>(G, rq, rk, temp, attn);
    attn_v<<<dim3(NSP / 64, HEADS), 96, 0, stream>>>(attn, qkvd, aout);
    gemm_bf16<false, true><<<dim3(NSP / 128, C_DIM / 64), 256, 0, stream>>>(
        wp, aout, (void*)ybuf, x, 192);

    // ---- feed-forward branch
    ln_kernel<<<NSP / 256, 256, 0, stream>>>(ybuf, ln2_w, ln2_b, lnb);
    gemm_bf16<true, false><<<dim3(NSP / 128, HF2P / 64), 256, 0, stream>>>(
        wi, lnb, (void*)hbuf, nullptr, 192);
    ffn_gate<<<dim3(16, T_DIM, HF), 256, 0, stream>>>(hbuf, ffn_dw, gbuf);
    gemm_bf16<false, true><<<dim3(NSP / 128, C_DIM / 64), 256, 0, stream>>>(
        wo, gbuf, (void*)outp, ybuf, HFP);
}
